// GTAN_14491219657222
// MI455X (gfx1250) — matrix operand order repaired and checked by reading, not compiled, not run on hardware
//
#include <hip/hip_runtime.h>

typedef __attribute__((ext_vector_type(16))) __bf16 v16bf;
typedef __attribute__((ext_vector_type(8)))  float  v8f;

#define NFEAT_IN 256
#define NHID     96
#define NOUT     64
#define HOPS     10

// ---------------- WMMA fragment loaders (ISA 7.12.2, wave32) ----------------
// A (16x32 bf16): lane l -> row m0+(l&15); lane group g=l>>4.
//   elem e : K = kk + 16g + e (contiguous, matches B)
__device__ __forceinline__ v16bf load_a_frag(const float* __restrict__ A, int lda,
                                             int m0, int kk) {
  const int lane = threadIdx.x & 31;
  const int g = lane >> 4;
  const int m = m0 + (lane & 15);
  const float* p = A + (size_t)m * lda + kk + 16 * g;
  v16bf a;
#pragma unroll
  for (int e = 0; e < 16; ++e) a[e] = (__bf16)p[e];
  return a;
}

// B (32x16 bf16): lane l -> col n0+(l&15); elem e -> K = kk + 16*(l>>4) + e.
__device__ __forceinline__ v16bf load_b_frag(const float* __restrict__ B, int ldb,
                                             int kk, int n0) {
  const int lane = threadIdx.x & 31;
  const int g = lane >> 4;
  const int n = n0 + (lane & 15);
  const float* p = B + (size_t)(kk + 16 * g) * ldb + n;
  v16bf b;
#pragma unroll
  for (int e = 0; e < 16; ++e) b[e] = (__bf16)p[(size_t)e * ldb];
  return b;
}

// ---------------- GEMM1: Xh = relu(X @ W1 + b1) ----------------
// grid.x = N/16 row tiles; block = 192 (6 waves, one 16-col tile each; 6*16=96)
__global__ void gtan_gemm1_relu(const float* __restrict__ X, const float* __restrict__ W,
                                const float* __restrict__ bias, float* __restrict__ Y) {
  const int m0 = blockIdx.x * 16;
  const int n0 = (threadIdx.x >> 5) * 16;
  v8f c = {};
#pragma unroll
  for (int kk = 0; kk < NFEAT_IN; kk += 32) {
    v16bf a = load_a_frag(X, NFEAT_IN, m0, kk);
    v16bf b = load_b_frag(W, NHID, kk, n0);
    c = __builtin_amdgcn_wmma_f32_16x16x32_bf16(false, a, false, b, (short)0, c,
                                                false, false);
  }
  const int lane = threadIdx.x & 31;
  const int g = lane >> 4;
  const int n = n0 + (lane & 15);
  const float bn = bias[n];
#pragma unroll
  for (int r = 0; r < 8; ++r) {
    const int m = m0 + r + 8 * g;
    float v = c[r] + bn;
    Y[(size_t)m * NHID + n] = v > 0.f ? v : 0.f;
  }
}

// ---------------- GEMM2: out = H @ W2 + b2 ----------------
// block = 128 (4 waves * 16 cols = 64)
__global__ void gtan_gemm2(const float* __restrict__ H, const float* __restrict__ W,
                           const float* __restrict__ bias, float* __restrict__ Y) {
  const int m0 = blockIdx.x * 16;
  const int n0 = (threadIdx.x >> 5) * 16;
  v8f c = {};
#pragma unroll
  for (int kk = 0; kk < NHID; kk += 32) {
    v16bf a = load_a_frag(H, NHID, m0, kk);
    v16bf b = load_b_frag(W, NOUT, kk, n0);
    c = __builtin_amdgcn_wmma_f32_16x16x32_bf16(false, a, false, b, (short)0, c,
                                                false, false);
  }
  const int lane = threadIdx.x & 31;
  const int g = lane >> 4;
  const int n = n0 + (lane & 15);
  const float bn = bias[n];
#pragma unroll
  for (int r = 0; r < 8; ++r) {
    const int m = m0 + r + 8 * g;
    Y[(size_t)m * NOUT + n] = c[r] + bn;
  }
}

// ---------------- per-node scalars on fixed x (once) ----------------
// x1[n] = x.a1 ; w2[n] = exp(lrelu(x.a1 + x.a2)). One wave per node.
__global__ void node_scalars_x(const float* __restrict__ Xh, const float* __restrict__ a1,
                               const float* __restrict__ a2, float* __restrict__ x1,
                               float* __restrict__ w2, int N) {
  const int node = blockIdx.x * 8 + (threadIdx.x >> 5);
  if (node >= N) return;
  const int lane = threadIdx.x & 31;
  const float* row = Xh + (size_t)node * NHID;
  float s1 = 0.f, s2 = 0.f;
#pragma unroll
  for (int r = 0; r < 3; ++r) {
    const int f = lane + 32 * r;
    const float v = row[f];
    s1 += v * a1[f];
    s2 += v * a2[f];
  }
#pragma unroll
  for (int off = 16; off > 0; off >>= 1) {
    s1 += __shfl_xor(s1, off, 32);
    s2 += __shfl_xor(s2, off, 32);
  }
  if (lane == 0) {
    const float v = s1 + s2;
    const float l = v >= 0.f ? v : 0.2f * v;
    x1[node] = s1;
    w2[node] = __expf(l);
  }
}

// h1[n] = h.a2 (per hop). One wave per node.
__global__ void node_scalar_h1(const float* __restrict__ H, const float* __restrict__ a2,
                               float* __restrict__ h1, int N) {
  const int node = blockIdx.x * 8 + (threadIdx.x >> 5);
  if (node >= N) return;
  const int lane = threadIdx.x & 31;
  const float* row = H + (size_t)node * NHID;
  float s = 0.f;
#pragma unroll
  for (int r = 0; r < 3; ++r) {
    const int f = lane + 32 * r;
    s += row[f] * a2[f];
  }
#pragma unroll
  for (int off = 16; off > 0; off >>= 1) s += __shfl_xor(s, off, 32);
  if (lane == 0) h1[node] = s;
}

// ---------------- init accumulators with self-loop term ----------------
__global__ void init_accum(const float* __restrict__ Xh, const float* __restrict__ w2,
                           float* __restrict__ num, float* __restrict__ divv, int total) {
  const int i = blockIdx.x * blockDim.x + threadIdx.x;
  if (i >= total) return;
  const int n = i / NHID;
  const float w = w2[n];
  num[i] = w * Xh[i];
  if (i - n * NHID == 0) divv[n] = w;
}

// ---------------- edge gather + scatter-add (the hot loop) ----------------
// One wave per edge; 96 features -> 3 per lane. Native f32 global atomics.
__global__ void edge_scatter(const int* __restrict__ s_idx, const int* __restrict__ t_idx,
                             const float* __restrict__ x1, const float* __restrict__ h1,
                             const float* __restrict__ H, float* __restrict__ num,
                             float* __restrict__ divv, int E) {
  const int e = blockIdx.x * 8 + (threadIdx.x >> 5);
  if (e >= E) return;
  const int lane = threadIdx.x & 31;
  const int s = s_idx[e];
  const int t = t_idx[e];
  const float v = x1[s] + h1[t];
  const float l = v >= 0.f ? v : 0.2f * v;
  const float w = __expf(l);
  if (lane == 0) unsafeAtomicAdd(&divv[s], w);
  const float* ht = H + (size_t)t * NHID;
  float* ns = num + (size_t)s * NHID;
#pragma unroll
  for (int r = 0; r < 3; ++r) {
    const int f = lane + 32 * r;
    unsafeAtomicAdd(&ns[f], w * ht[f]);
  }
}

// ---------------- h = elu(num / div), in place ----------------
__global__ void elu_norm(float* __restrict__ num, const float* __restrict__ divv,
                         int total) {
  const int i = blockIdx.x * blockDim.x + threadIdx.x;
  if (i >= total) return;
  const int n = i / NHID;
  const float v = num[i] / divv[n];
  num[i] = v > 0.f ? v : __expf(v) - 1.f;
}

// ---------------- host-side orchestration ----------------
extern "C" void kernel_launch(void* const* d_in, const int* in_sizes, int n_in,
                              void* d_out, int out_size, void* d_ws, size_t ws_size,
                              hipStream_t stream) {
  const float* x  = (const float*)d_in[0];
  const int*   ei = (const int*)d_in[1];
  const float* W1 = (const float*)d_in[2];
  const float* b1 = (const float*)d_in[3];
  const float* a1 = (const float*)d_in[4];
  const float* a2 = (const float*)d_in[5];
  const float* W2 = (const float*)d_in[6];
  const float* b2 = (const float*)d_in[7];

  const int N = in_sizes[0] / NFEAT_IN;   // 50000
  const int E = in_sizes[1] / 2;          // 800000
  const int* s_idx = ei;                  // edge_index[0]
  const int* t_idx = ei + E;              // edge_index[1]

  char* ws = (char*)d_ws;
  size_t off = 0;
  auto grab = [&](size_t bytes) -> void* {
    void* p = ws + off;
    off = (off + bytes + 255) & ~(size_t)255;
    return p;
  };
  float* Xh   = (float*)grab((size_t)N * NHID * sizeof(float));
  float* bufA = (float*)grab((size_t)N * NHID * sizeof(float));
  float* bufB = (float*)grab((size_t)N * NHID * sizeof(float));
  float* x1   = (float*)grab((size_t)N * sizeof(float));
  float* w2   = (float*)grab((size_t)N * sizeof(float));
  float* h1   = (float*)grab((size_t)N * sizeof(float));
  float* divv = (float*)grab((size_t)N * sizeof(float));
  (void)ws_size; (void)n_in; (void)out_size;

  const int rowTiles = N / 16;            // 3125 (exact)
  const int total = N * NHID;

  // Layer 1: WMMA GEMM + bias + relu
  gtan_gemm1_relu<<<rowTiles, 192, 0, stream>>>(x, W1, b1, Xh);
  // x-dependent scalars are hop-invariant: compute once
  node_scalars_x<<<(N + 7) / 8, 256, 0, stream>>>(Xh, a1, a2, x1, w2, N);

  const float* h = Xh;
  for (int hop = 0; hop < HOPS; ++hop) {
    float* num = (hop & 1) ? bufB : bufA;
    node_scalar_h1<<<(N + 7) / 8, 256, 0, stream>>>(h, a2, h1, N);
    init_accum<<<(total + 255) / 256, 256, 0, stream>>>(Xh, w2, num, divv, total);
    edge_scatter<<<(E + 7) / 8, 256, 0, stream>>>(s_idx, t_idx, x1, h1, h, num, divv, E);
    elu_norm<<<(total + 255) / 256, 256, 0, stream>>>(num, divv, total);
    h = num;  // elu'd num is next hop's h
  }

  // Layer 2: WMMA GEMM + bias -> output
  gtan_gemm2<<<rowTiles, 128, 0, stream>>>(h, W2, b2, (float*)d_out);
}
